// AlgorithmExecutor_54786602827887
// MI455X (gfx1250) — matrix operand order repaired and checked by reading, not compiled, not run on hardware
//
#include <hip/hip_runtime.h>

// GNN BFS executor for MI455X (gfx1250, wave32, WMMA).
//
// Algebraic refactor: msg = relu([h_dst|h_src|e] @ Wm + b)
//                         = relu(P[dst,:128] + P[src,128:] + e*wm_e + b)
// with P = h @ [Wm_dst | Wm_src]  (a NODE gemm, 1.3 GFLOP, not a 42 GFLOP edge gemm).
// GEMMs run on v_wmma_f32_16x16x32_f16 (fully unrolled, distinct B fragments per
// column tile so loads pipeline past WMMAs); the edge pass is pure L2-resident
// gather + atomic-max (P ~20MB, aggr ~10MB << 192MB L2).

typedef __attribute__((ext_vector_type(16))) _Float16 v16h;
typedef __attribute__((ext_vector_type(8)))  _Float16 v8h;
typedef __attribute__((ext_vector_type(8)))  float    v8f;

union Frag16 { v16h v; v8h h[2]; };

#define HDIM 128
#define NLAYERS 3

// ---------------------------------------------------------------- utilities
__global__ void zero_f32_kernel(float* __restrict__ p, long n) {
  long i = (long)blockIdx.x * blockDim.x + threadIdx.x;
  if (i < n) p[i] = 0.0f;
}

// msg_w [257,128] f32 -> Wm [128,256] f16 : col<128 from rows 0..127 (dst part),
// col>=128 from rows 128..255 (src part).
__global__ void cvt_msg_w_kernel(const float* __restrict__ msg_w, _Float16* __restrict__ Wm) {
  int i = blockIdx.x * blockDim.x + threadIdx.x;
  if (i >= 128 * 256) return;
  int k = i >> 8, n = i & 255;
  float v = (n < 128) ? msg_w[k * 128 + n] : msg_w[(128 + k) * 128 + (n - 128)];
  Wm[k * 256 + n] = (_Float16)v;
}

// upd_w [256,128] f32 -> f16 (same layout)
__global__ void cvt_upd_w_kernel(const float* __restrict__ upd_w, _Float16* __restrict__ Wu) {
  int i = blockIdx.x * blockDim.x + threadIdx.x;
  if (i < 256 * 128) Wu[i] = (_Float16)upd_w[i];
}

// h = x*enc_w + enc_b ; write f32 copy and f16 copy into X16[:, 0:128] (ld 256)
__global__ void encoder_kernel(const float* __restrict__ x, const float* __restrict__ enc_w,
                               const float* __restrict__ enc_b, float* __restrict__ hf32,
                               _Float16* __restrict__ X16, int N, int Npad) {
  long i = (long)blockIdx.x * blockDim.x + threadIdx.x;
  if (i >= (long)Npad * HDIM) return;
  int n = (int)(i >> 7), f = (int)(i & 127);
  float xv = (n < N) ? x[n] : 0.0f;
  float h = xv * enc_w[f] + enc_b[f];
  hf32[(size_t)n * HDIM + f] = h;
  X16[(size_t)n * 256 + f] = (_Float16)h;
}

// ---------------------------------------------------------------- WMMA GEMM
// C[M, Ncols] = X[M, K](f16, ldx) @ W[K, Ncols](f16, ldw) [+ bias][, relu]
// One wave -> 16x64 tile. blockDim = 128 (4 waves -> 64 rows). grid = (M/64, Ncols/64).
// K is a template constant (128 or 256) so the whole K loop unrolls and each
// column tile owns distinct B fragment registers -> loads of step k+1 overlap
// the WMMAs of step k instead of a serialized load/wait0/wmma chain.
template <int K>
__global__ __launch_bounds__(128)
void wmma_gemm_kernel(const _Float16* __restrict__ X, int ldx,
                      const _Float16* __restrict__ W, int ldw,
                      float* __restrict__ Cf, int ldc,
                      _Float16* __restrict__ C16, int ldc16,
                      const float* __restrict__ bias, int relu) {
  const int wave = threadIdx.x >> 5;
  const int lane = threadIdx.x & 31;
  const int lhi  = lane >> 4;     // 0/1: lane half
  const int llo  = lane & 15;
  const int rowTile = blockIdx.x * 64 + wave * 16;
  const int colBase = blockIdx.y * 64;

  v8f acc[4] = {};

  // A fragment (16x32 f16): lanes 0-15 -> M=0..15; VGPR0-3 hold K = lhi*8 + 0..7,
  // VGPR4-7 hold K = 16 + lhi*8 + 0..7  ==> two contiguous b128 loads per lane.
  const _Float16* aPtr = X + (size_t)(rowTile + llo) * ldx + lhi * 8;
  // B fragment (32x16 f16): lane holds column N = llo; element s holds K = lhi*8 + s (s<8), 16 + lhi*8 + (s-8) (s>=8).
  const _Float16* bPtr = W + colBase + llo;

#pragma unroll
  for (int k0 = 0; k0 < K; k0 += 32) {
    Frag16 a;
    a.h[0] = *(const v8h*)(aPtr + k0);
    a.h[1] = *(const v8h*)(aPtr + k0 + 16);
    const _Float16* bk = bPtr + (size_t)k0 * ldw;
    Frag16 b0, b1, b2, b3;                     // distinct regs per column tile
#pragma unroll
    for (int s = 0; s < 16; ++s) {
      const int kk = (s < 8) ? (lhi * 8 + s) : (16 + lhi * 8 + (s - 8));
      const _Float16* br = bk + (size_t)kk * ldw;
      b0.v[s] = br[0]; b1.v[s] = br[16]; b2.v[s] = br[32]; b3.v[s] = br[48];
    }
    acc[0] = __builtin_amdgcn_wmma_f32_16x16x32_f16(false, a.v, false, b0.v,
                                                    (short)0, acc[0], false, false);
    acc[1] = __builtin_amdgcn_wmma_f32_16x16x32_f16(false, a.v, false, b1.v,
                                                    (short)0, acc[1], false, false);
    acc[2] = __builtin_amdgcn_wmma_f32_16x16x32_f16(false, a.v, false, b2.v,
                                                    (short)0, acc[2], false, false);
    acc[3] = __builtin_amdgcn_wmma_f32_16x16x32_f16(false, a.v, false, b3.v,
                                                    (short)0, acc[3], false, false);
  }

  // C layout: VGPR j -> M = j + 8*lhi, N = llo (per 16x16 f32 accumulator table)
#pragma unroll
  for (int t = 0; t < 4; ++t) {
    int col = colBase + t * 16 + llo;
    float bv = bias ? bias[col] : 0.0f;
#pragma unroll
    for (int j = 0; j < 8; ++j) {
      int row = rowTile + lhi * 8 + j;
      float v = acc[t][j] + bv;
      if (relu) v = fmaxf(v, 0.0f);
      Cf[(size_t)row * ldc + col] = v;
      if (C16) C16[(size_t)row * ldc16 + col] = (_Float16)v;
    }
  }
}

// ---------------------------------------------------------------- edge pass
// msg[e,f] = relu(P[dst,f] + P[src,128+f] + e_attr[e]*msg_w[256,f] + msg_b[f])
// scatter-max into aggr[dst,f] (float-as-uint atomicMax: all values >= 0).
__global__ void edge_msg_kernel(const long long* __restrict__ ei,  // [2,E] int64
                                const float* __restrict__ eattr,
                                const float* __restrict__ P,       // [Npad,256]
                                const float* __restrict__ msg_w,   // [257,128]
                                const float* __restrict__ msg_b,
                                float* __restrict__ aggr, int E) {
  long i = (long)blockIdx.x * blockDim.x + threadIdx.x;
  if (i >= (long)E * HDIM) return;
  int e = (int)(i >> 7);
  int f = (int)(i & 127);
  int src = (int)ei[e];                 // edge_index[0] = j (source)
  int dst = (int)ei[(size_t)E + e];     // edge_index[1] = i (target)
  float m = P[(size_t)dst * 256 + f] + P[(size_t)src * 256 + 128 + f]
          + eattr[e] * msg_w[256 * HDIM + f] + msg_b[f];
  m = fmaxf(m, 0.0f);
  if (m > 0.0f)
    atomicMax((unsigned int*)(aggr + (size_t)dst * HDIM + f), __float_as_uint(m));
}

// aggr (f32, >=0) -> f16 into X16[:, 128:256]
__global__ void aggr_cvt_kernel(const float* __restrict__ aggr, _Float16* __restrict__ X16,
                                long total) {
  long i = (long)blockIdx.x * blockDim.x + threadIdx.x;
  if (i >= total) return;
  long n = i >> 7; int f = (int)(i & 127);
  X16[(size_t)n * 256 + 128 + f] = (_Float16)aggr[i];
}

// ---------------------------------------------------------------- heads
// out[n] = sigmoid(dot(h[n,:], dec_w) + dec_b) ; one wave32 per node
__global__ void decode_kernel(const float* __restrict__ h, const float* __restrict__ dec_w,
                              const float* __restrict__ dec_b, float* __restrict__ out, int N) {
  int warp = (int)(((long)blockIdx.x * blockDim.x + threadIdx.x) >> 5);
  int lane = threadIdx.x & 31;
  if (warp >= N) return;
  float s = 0.0f;
#pragma unroll
  for (int j = 0; j < 4; ++j) {
    int f = lane + j * 32;
    s += h[(size_t)warp * HDIM + f] * dec_w[f];
  }
#pragma unroll
  for (int off = 16; off > 0; off >>= 1) s += __shfl_down(s, off, 32);
  if (lane == 0) out[warp] = 1.0f / (1.0f + __expf(-(s + dec_b[0])));
}

// per-feature column sums of h over nodes (coalesced; one atomicAdd per thread per block)
__global__ __launch_bounds__(128)
void feat_sum_kernel(const float* __restrict__ h, float* __restrict__ hsum, int N) {
  int f = threadIdx.x;
  int base = blockIdx.x * 256;
  float s = 0.0f;
  for (int r = 0; r < 256; ++r) {
    int n = base + r;
    if (n < N) s += h[(size_t)n * HDIM + f];
  }
  atomicAdd(hsum + f, s);
}

__global__ __launch_bounds__(128)
void term_kernel(const float* __restrict__ hsum, const float* __restrict__ term_w,
                 const float* __restrict__ term_b, float* __restrict__ out, int N) {
  __shared__ float red[128];
  int f = threadIdx.x;
  red[f] = (hsum[f] / (float)N) * term_w[f];
  __syncthreads();
  for (int s = 64; s > 0; s >>= 1) {
    if (f < s) red[f] += red[f + s];
    __syncthreads();
  }
  if (f == 0) out[0] = 1.0f / (1.0f + __expf(-(red[0] + term_b[0])));
}

// ---------------------------------------------------------------- launcher
extern "C" void kernel_launch(void* const* d_in, const int* in_sizes, int n_in,
                              void* d_out, int out_size, void* d_ws, size_t ws_size,
                              hipStream_t stream) {
  const float*      x      = (const float*)d_in[0];
  const long long*  ei     = (const long long*)d_in[1];  // reference dtype is int64
  const float*      eattr  = (const float*)d_in[2];
  const float*      enc_w  = (const float*)d_in[3];
  const float*      enc_b  = (const float*)d_in[4];
  const float*      msg_w  = (const float*)d_in[5];
  const float*      msg_b  = (const float*)d_in[6];
  const float*      upd_w  = (const float*)d_in[7];
  const float*      upd_b  = (const float*)d_in[8];
  const float*      dec_w  = (const float*)d_in[9];
  const float*      dec_b  = (const float*)d_in[10];
  const float*      term_w = (const float*)d_in[11];
  const float*      term_b = (const float*)d_in[12];
  float* out = (float*)d_out;   // [N] node outputs, then [1] termination

  const int N    = in_sizes[0];            // 20000
  const int E    = in_sizes[2];            // 640000 (edge_attr flat count)
  const int Npad = ((N + 63) / 64) * 64;   // full 64-row WMMA blocks, no tail guards

  // ---- carve workspace (all 256B-aligned)
  char* ws = (char*)d_ws;
  size_t off = 0;
  auto carve = [&](size_t bytes) -> char* {
    char* p = ws + off;
    off = (off + bytes + 255) & ~(size_t)255;
    return p;
  };
  _Float16* X16a = (_Float16*)carve((size_t)Npad * 256 * sizeof(_Float16)); // [h|aggr] f16
  _Float16* X16b = (_Float16*)carve((size_t)Npad * 256 * sizeof(_Float16)); // ping-pong
  float*    hf32 = (float*)   carve((size_t)Npad * HDIM * sizeof(float));
  float*    P    = (float*)   carve((size_t)Npad * 256 * sizeof(float));
  float*    aggr = (float*)   carve((size_t)Npad * HDIM * sizeof(float));
  _Float16* Wm   = (_Float16*)carve((size_t)128 * 256 * sizeof(_Float16));
  _Float16* Wu   = (_Float16*)carve((size_t)256 * 128 * sizeof(_Float16));
  float*    hsum = (float*)   carve(128 * sizeof(float));
  (void)ws_size; (void)n_in; (void)out_size;

  const int TB = 256;

  // weights -> f16 (msg gets [dst|src] column-concat layout)
  cvt_msg_w_kernel<<<(128 * 256 + TB - 1) / TB, TB, 0, stream>>>(msg_w, Wm);
  cvt_upd_w_kernel<<<(256 * 128 + TB - 1) / TB, TB, 0, stream>>>(upd_w, Wu);

  // encoder
  long nh = (long)Npad * HDIM;
  encoder_kernel<<<(int)((nh + TB - 1) / TB), TB, 0, stream>>>(x, enc_w, enc_b, hf32, X16a, N, Npad);
  zero_f32_kernel<<<1, 128, 0, stream>>>(hsum, 128);

  dim3 gemm_block(128);
  _Float16* cur = X16a;
  _Float16* nxt = X16b;
  for (int layer = 0; layer < NLAYERS; ++layer) {
    // aggr = 0 (PyG zero-init semantics; messages are >= 0)
    zero_f32_kernel<<<(int)((nh + TB - 1) / TB), TB, 0, stream>>>(aggr, nh);

    // P[N,256] = h @ [Wm_dst | Wm_src]   (K=128)
    wmma_gemm_kernel<128><<<dim3(Npad / 64, 256 / 64), gemm_block, 0, stream>>>(
        cur, 256, Wm, 256, P, 256, (_Float16*)nullptr, 0, (const float*)nullptr, 0);

    // edge pass: add parts + bias + relu, scatter-max into aggr
    long ne = (long)E * HDIM;
    edge_msg_kernel<<<(int)((ne + TB - 1) / TB), TB, 0, stream>>>(ei, eattr, P, msg_w, msg_b, aggr, E);

    // aggr -> f16 into cur[:, 128:256]
    aggr_cvt_kernel<<<(int)((nh + TB - 1) / TB), TB, 0, stream>>>(aggr, cur, nh);

    // h_new = relu([h|aggr] @ upd_w + upd_b)  (K=256) -> hf32 and nxt[:, 0:128] f16
    wmma_gemm_kernel<256><<<dim3(Npad / 64, 128 / 64), gemm_block, 0, stream>>>(
        cur, 256, Wu, 128, hf32, 128, nxt, 256, upd_b, 1);

    _Float16* t = cur; cur = nxt; nxt = t;
  }

  // heads
  decode_kernel<<<(N * 32 + TB - 1) / TB, TB, 0, stream>>>(hf32, dec_w, dec_b, out, N);
  feat_sum_kernel<<<(N + 255) / 256, 128, 0, stream>>>(hf32, hsum, N);
  term_kernel<<<1, 128, 0, stream>>>(hsum, term_w, term_b, out + N, N);
}
